// DrugEncoder_60945585931020
// MI455X (gfx1250) — matrix operand order repaired and checked by reading, not compiled, not run on hardware
//
#include <hip/hip_runtime.h>
#include <hip/hip_bf16.h>
#include <stdint.h>

// CDNA5 WMMA vector types
typedef __attribute__((ext_vector_type(16))) _Float16 v16h;
typedef __attribute__((ext_vector_type(8)))  _Float16 v8h;
typedef __attribute__((ext_vector_type(8)))  float    v8f;

#define HID 128

// ---------------------------------------------------------------------------
// degree = segment_sum(ones, col) ; then deg <- rsqrt(deg + 1)
// ---------------------------------------------------------------------------
__global__ void gcn_deg_count(const int* __restrict__ col,
                              float* __restrict__ deg, int n_edges) {
  int t = blockIdx.x * blockDim.x + threadIdx.x;
  if (t < n_edges) atomicAdd(&deg[col[t]], 1.0f);
}

__global__ void gcn_deg_finalize(float* __restrict__ deg, int n_nodes) {
  int t = blockIdx.x * blockDim.x + threadIdx.x;
  if (t < n_nodes) deg[t] = rsqrtf(deg[t] + 1.0f);
}

// ---------------------------------------------------------------------------
// Wt[col][k] = (f16) W[k][col], zero-padded to K=128 so GEMM B-fragments are
// contiguous 32B loads per lane.
// ---------------------------------------------------------------------------
__global__ void gcn_convert_wt(const float* __restrict__ W,
                               _Float16* __restrict__ Wt, int K) {
  int t = blockIdx.x * blockDim.x + threadIdx.x;
  if (t >= HID * HID) return;
  int c = t >> 7;       // output column (0..127)
  int k = t & 127;      // padded K index
  float v = (k < K) ? W[k * HID + c] : 0.0f;
  Wt[c * HID + k] = (_Float16)v;
}

// ---------------------------------------------------------------------------
// Pack f32 [N,K] row-major into zero-padded f16 [N,KPAD]
// ---------------------------------------------------------------------------
__global__ void gcn_pack_a_f16(const float* __restrict__ X,
                               _Float16* __restrict__ A16,
                               int n_nodes, int K, int KPAD) {
  long long t = (long long)blockIdx.x * blockDim.x + threadIdx.x;
  if (t >= (long long)n_nodes * KPAD) return;
  int k = (int)(t % KPAD);
  long long node = t / KPAD;
  float v = (k < K) ? X[node * K + k] : 0.0f;
  A16[t] = (_Float16)v;
}

// ---------------------------------------------------------------------------
// HW[N,128] = A16[N,KPAD] @ W[K,128]   via v_wmma_f32_16x16x32_f16
// One wave computes a 16-row stripe across all 128 columns (8 WMMA tiles),
// fully unrolled over K. A and B fragments are aligned b128 loads.
// ---------------------------------------------------------------------------
template <int KPAD>
__global__ __launch_bounds__(256)
void gcn_gemm_wmma(const _Float16* __restrict__ A16,
                   const _Float16* __restrict__ Wt,
                   float* __restrict__ HW,
                   int n_nodes) {
  const int wave = (blockIdx.x * blockDim.x + threadIdx.x) >> 5;
  const int lane = threadIdx.x & 31;
  const long long rowBase = (long long)wave * 16;
  if (rowBase >= n_nodes) return;                 // whole wave exits: EXEC stays full

  const int half = lane >> 4;                     // 0: lanes 0-15, 1: lanes 16-31
  const int m    = lane & 15;
  long long arow = rowBase + m;
  if (arow >= n_nodes) arow = n_nodes - 1;        // clamp loads; stores are guarded
  const _Float16* __restrict__ ap = A16 + arow * (long long)KPAD;

  v8f acc[8];
#pragma unroll
  for (int t = 0; t < 8; ++t) acc[t] = (v8f){0.f,0.f,0.f,0.f,0.f,0.f,0.f,0.f};

#pragma unroll
  for (int kk = 0; kk < KPAD; kk += 32) {
    // A fragment: lanes 0-15 hold row m, K = kk+{0..15};
    //             lanes 16-31 hold row m, K = kk+{16..31}.
    v8h alo = *(const v8h*)(ap + kk + half * 16);
    v8h ahi = *(const v8h*)(ap + kk + half * 16 + 8);
    v16h a = __builtin_shufflevector(alo, ahi,
                                     0, 1, 2, 3, 4, 5, 6, 7,
                                     8, 9, 10, 11, 12, 13, 14, 15);
#pragma unroll
    for (int t = 0; t < 8; ++t) {
      // B fragment: lanes 0-15 = col n, K kk..kk+15 ; lanes 16-31 same cols, K kk+16..kk+31
      v16h b = *(const v16h*)(Wt + ((t * 16 + m) * HID + kk + half * 16));
      acc[t] = __builtin_amdgcn_wmma_f32_16x16x32_f16(
          /*neg_a=*/false, a, /*neg_b=*/false, b,
          /*c_mod=*/(short)0, acc[t], /*reuse_a=*/false, /*reuse_b=*/false);
    }
  }

  // C/D layout: lane gives column n = m ; VGPR r gives row half*8 + r.
#pragma unroll
  for (int t = 0; t < 8; ++t) {
#pragma unroll
    for (int r = 0; r < 8; ++r) {
      long long row = rowBase + half * 8 + r;
      if (row < (long long)n_nodes)
        HW[row * HID + t * 16 + m] = acc[t][r];
    }
  }
}

// ---------------------------------------------------------------------------
// AGG[i,c] = HW[i,c] * dinv[i]^2   (self-loop term; full overwrite)
// ---------------------------------------------------------------------------
__global__ void gcn_init_agg(const float* __restrict__ HW,
                             const float* __restrict__ dinv,
                             float* __restrict__ AGG, int n_nodes) {
  long long t = (long long)blockIdx.x * blockDim.x + threadIdx.x;
  if (t >= (long long)n_nodes * HID) return;
  int node = (int)(t >> 7);
  float d = dinv[node];
  AGG[t] = HW[t] * d * d;
}

// ---------------------------------------------------------------------------
// For each edge e: AGG[col[e], :] += HW[row[e], :] * dinv[row]*dinv[col]
// One lane per (edge, channel) -> coalesced 4B atomics.
// ---------------------------------------------------------------------------
__global__ void gcn_edge_scatter(const int* __restrict__ row,
                                 const int* __restrict__ col,
                                 const float* __restrict__ dinv,
                                 const float* __restrict__ HW,
                                 float* __restrict__ AGG, int n_edges) {
  long long t = (long long)blockIdx.x * blockDim.x + threadIdx.x;
  if (t >= (long long)n_edges * HID) return;
  int e = (int)(t >> 7);
  int c = (int)(t & 127);
  int r  = row[e];
  int cl = col[e];
  float norm = dinv[r] * dinv[cl];
  atomicAdd(&AGG[(long long)cl * HID + c], HW[(long long)r * HID + c] * norm);
}

// ---------------------------------------------------------------------------
// v = relu(AGG[i,c] + b[c]); AGG <- v (f32, exact path for pooling)
//                            A16 <- v (f16, next layer's GEMM input)
// ---------------------------------------------------------------------------
__global__ void gcn_bias_relu(float* __restrict__ AGG,
                              const float* __restrict__ bias,
                              _Float16* __restrict__ A16, int n_nodes) {
  long long t = (long long)blockIdx.x * blockDim.x + threadIdx.x;
  if (t >= (long long)n_nodes * HID) return;
  float v = AGG[t] + bias[(int)(t & 127)];
  v = v > 0.0f ? v : 0.0f;
  AGG[t] = v;
  A16[t] = (_Float16)v;
}

// ---------------------------------------------------------------------------
// Mean pool over batch ids
// ---------------------------------------------------------------------------
__global__ void gcn_pool_count(const int* __restrict__ batch,
                               float* __restrict__ counts, int n_nodes) {
  int t = blockIdx.x * blockDim.x + threadIdx.x;
  if (t < n_nodes) atomicAdd(&counts[batch[t]], 1.0f);
}

__global__ void gcn_pool_sum(const int* __restrict__ batch,
                             const float* __restrict__ H,
                             float* __restrict__ out, int n_nodes) {
  long long t = (long long)blockIdx.x * blockDim.x + threadIdx.x;
  if (t >= (long long)n_nodes * HID) return;
  int node = (int)(t >> 7);
  int c = (int)(t & 127);
  atomicAdd(&out[(long long)batch[node] * HID + c], H[t]);
}

__global__ void gcn_pool_div(float* __restrict__ out,
                             const float* __restrict__ counts, int n_graphs) {
  int t = blockIdx.x * blockDim.x + threadIdx.x;
  if (t >= n_graphs * HID) return;
  out[t] /= fmaxf(counts[t >> 7], 1.0f);
}

// ---------------------------------------------------------------------------

static inline unsigned cdiv(long long a, long long b) { return (unsigned)((a + b - 1) / b); }

extern "C" void kernel_launch(void* const* d_in, const int* in_sizes, int n_in,
                              void* d_out, int out_size, void* d_ws, size_t ws_size,
                              hipStream_t stream) {
  // inputs per reference order
  const float* x     = (const float*)d_in[0];     // [N, 78]
  const int*   ei    = (const int*)d_in[1];       // [2, E]
  const int*   batch = (const int*)d_in[2];       // [N]
  const float* W1 = (const float*)d_in[3];
  const float* b1 = (const float*)d_in[4];
  const float* W2 = (const float*)d_in[5];
  const float* b2 = (const float*)d_in[6];
  const float* W3 = (const float*)d_in[7];
  const float* b3 = (const float*)d_in[8];
  (void)n_in; (void)ws_size;

  const int n_nodes  = in_sizes[2];
  const int n_edges  = in_sizes[1] / 2;
  const int node_dim = in_sizes[0] / n_nodes;     // 78
  const int n_graphs = out_size / HID;            // 16384
  const int* erow = ei;
  const int* ecol = ei + n_edges;

  const int KPAD1 = ((node_dim + 31) / 32) * 32;  // 78 -> 96

  // workspace carve-up
  char* p = (char*)d_ws;
  float* deg = (float*)p;            p += (size_t)n_nodes * sizeof(float);   // becomes dinv
  float* counts = (float*)p;         p += (size_t)n_graphs * sizeof(float);
  _Float16* Wt = (_Float16*)p;       p += (size_t)3 * HID * HID * sizeof(_Float16);
  uintptr_t up = ((uintptr_t)p + 255) & ~(uintptr_t)255;
  _Float16* A16 = (_Float16*)up;     up += (size_t)n_nodes * HID * sizeof(_Float16);
  up = (up + 255) & ~(uintptr_t)255;
  float* HW  = (float*)up;
  float* AGG = HW + (size_t)n_nodes * HID;

  const int BT = 256;
  const long long NH = (long long)n_nodes * HID;
  const long long EH = (long long)n_edges * HID;

  // degree normalization
  hipMemsetAsync(deg, 0, (size_t)n_nodes * sizeof(float), stream);
  gcn_deg_count<<<cdiv(n_edges, BT), BT, 0, stream>>>(ecol, deg, n_edges);
  gcn_deg_finalize<<<cdiv(n_nodes, BT), BT, 0, stream>>>(deg, n_nodes);

  // f16 transposed weights (zero-padded K -> 128)
  gcn_convert_wt<<<cdiv(HID * HID, BT), BT, 0, stream>>>(W1, Wt + 0 * HID * HID, node_dim);
  gcn_convert_wt<<<cdiv(HID * HID, BT), BT, 0, stream>>>(W2, Wt + 1 * HID * HID, HID);
  gcn_convert_wt<<<cdiv(HID * HID, BT), BT, 0, stream>>>(W3, Wt + 2 * HID * HID, HID);

  // pack x -> f16 [N, 96]
  gcn_pack_a_f16<<<cdiv((long long)n_nodes * KPAD1, BT), BT, 0, stream>>>(
      x, A16, n_nodes, node_dim, KPAD1);

  const long long n_row_tiles = ((long long)n_nodes + 15) / 16;   // waves needed
  const unsigned gemm_blocks = cdiv(n_row_tiles * 32, BT);

  const float* biases[3] = {b1, b2, b3};
  for (int l = 0; l < 3; ++l) {
    if (l == 0)
      gcn_gemm_wmma<96><<<gemm_blocks, BT, 0, stream>>>(A16, Wt, HW, n_nodes);
    else
      gcn_gemm_wmma<128><<<gemm_blocks, BT, 0, stream>>>(A16, Wt + (size_t)l * HID * HID,
                                                         HW, n_nodes);
    gcn_init_agg<<<cdiv(NH, BT), BT, 0, stream>>>(HW, deg, AGG, n_nodes);
    gcn_edge_scatter<<<cdiv(EH, BT), BT, 0, stream>>>(erow, ecol, deg, HW, AGG, n_edges);
    gcn_bias_relu<<<cdiv(NH, BT), BT, 0, stream>>>(AGG, biases[l], A16, n_nodes);
  }

  // global mean pool (from exact f32 activations)
  hipMemsetAsync(d_out, 0, (size_t)out_size * sizeof(float), stream);
  hipMemsetAsync(counts, 0, (size_t)n_graphs * sizeof(float), stream);
  gcn_pool_count<<<cdiv(n_nodes, BT), BT, 0, stream>>>(batch, counts, n_nodes);
  gcn_pool_sum<<<cdiv(NH, BT), BT, 0, stream>>>(batch, AGG, (float*)d_out, n_nodes);
  gcn_pool_div<<<cdiv((long long)n_graphs * HID, BT), BT, 0, stream>>>((float*)d_out, counts, n_graphs);
}
